// BernoulliEdge_7438883357344
// MI455X (gfx1250) — matrix operand order repaired and checked by reading, not compiled, not run on hardware
//
#include <hip/hip_runtime.h>

// ---------- constants (fixed by reference: B=64, N=512, D=512) ----------
#define BB   64
#define NN   512
#define DD   512
#define K2D  1024                 // 2*D, GEMM K dimension
#define NEGV -1000000000.0f

typedef __attribute__((ext_vector_type(16))) __bf16 v16bf;
typedef __attribute__((ext_vector_type(8)))  float  v8f;

__device__ __forceinline__ unsigned short f2bf(float x) {
    union { float f; unsigned u; } c; c.f = x;
    return (unsigned short)(c.u >> 16);
}

// hash -> uniform in (0,1)
__device__ __forceinline__ float urand(unsigned e, unsigned b, unsigned i, unsigned j) {
    unsigned x = 0x9E3779B9u * (e + 1u) ^ 0x85EBCA6Bu * b ^ 0xC2B2AE35u * i
               ^ 0x27D4EB2Fu * j ^ 0x2A5F3C1Du;
    x ^= x >> 16; x *= 0x7FEB352Du;
    x ^= x >> 15; x *= 0x846CA68Bu;
    x ^= x >> 16;
    return ((float)((x >> 8) + 1u)) * (1.0f / 16777217.0f);   // strictly in (0,1)
}

// ====================================================================
// Kernel 1: repack W1 (1024x512 f32, row-major [k][n]) into the WMMA
// B-operand bf16 layout: packed[((nt*32+ks)*32+lane)*16 + j]
//   lane<16 : K = ks*32 + j       , col = nt*16 + lane
//   lane>=16: K = ks*32 + 16 + j  , col = nt*16 + (lane-16)
// so each lane's 16 bf16 (32 bytes) are contiguous in memory.
// ====================================================================
__global__ void w1pack_kernel(const float* __restrict__ W1,
                              unsigned short* __restrict__ W1p) {
    int tid  = blockIdx.x * 256 + threadIdx.x;    // 0 .. 524287
    int j    = tid & 15;
    int lane = (tid >> 4) & 31;
    int ks   = (tid >> 9) & 31;
    int nt   = tid >> 14;                         // 0..31
    int K    = ks * 32 + ((lane & 16) ? 16 : 0) + j;
    int col  = nt * 16 + (lane & 15);
    W1p[tid] = f2bf(W1[K * DD + col]);
}

// ====================================================================
// Kernel 2: fused  logits = tanh(net_in @ W1 + b1) @ W2 + b2
// One workgroup (256 thr = 8 wave32) per 16 rows of net_in.
// A tile (16 x 1024 bf16) staged in LDS; each wave owns 4 n-tiles.
// ====================================================================
__global__ void __launch_bounds__(256)
logits_kernel(const float* __restrict__ nodes,
              const int*   __restrict__ num_nodes,
              const unsigned short* __restrict__ W1p,
              const float* __restrict__ b1,
              const float* __restrict__ W2,
              const float* __restrict__ b2,
              float* __restrict__ logits) {
    __shared__ __align__(16) unsigned short smA[16 * 1032];   // padded row stride
    __shared__ float lpart[16];

    const int g0 = blockIdx.x * 16;          // first global row (b*512 + n)
    const int bb = g0 >> 9;
    const int nn = num_nodes[bb];

    if (threadIdx.x < 16) lpart[threadIdx.x] = 0.0f;

    // ---- stage A: 16 rows x 1024 bf16 (convert f32 -> bf16 on the fly) ----
    // 2048 chunks of 8 elements; 8 chunks per thread.
    for (int c = threadIdx.x; c < 2048; c += 256) {
        int m   = c >> 7;                    // 0..15
        int k8  = (c & 127) << 3;            // 0,8,...,1016
        int row = (g0 + m) & 511;
        const float* src = (k8 < DD)
            ? nodes + ((size_t)(bb * NN + nn) * DD + k8)          // curr half
            : nodes + ((size_t)(bb * NN + row) * DD + (k8 - DD)); // nodes half
        float4 f0 = *(const float4*)(src);
        float4 f1 = *(const float4*)(src + 4);
        uint4 o;
        o.x = (unsigned)f2bf(f0.x) | ((unsigned)f2bf(f0.y) << 16);
        o.y = (unsigned)f2bf(f0.z) | ((unsigned)f2bf(f0.w) << 16);
        o.z = (unsigned)f2bf(f1.x) | ((unsigned)f2bf(f1.y) << 16);
        o.w = (unsigned)f2bf(f1.z) | ((unsigned)f2bf(f1.w) << 16);
        *(uint4*)&smA[m * 1032 + k8] = o;
    }
    __syncthreads();

    const int wave = threadIdx.x >> 5;
    const int lane = threadIdx.x & 31;
    const int mrow = lane & 15;              // A row this lane supplies
    const int koff = (lane >> 4) << 3;       // 0 or 8; A uses contiguous k = 16h + s (via 2*koff)

    float part[8];
#pragma unroll
    for (int r = 0; r < 8; ++r) part[r] = 0.0f;

    const v16bf* wp = (const v16bf*)W1p;

    for (int t = 0; t < 4; ++t) {
        int nt = wave + t * 8;               // n-tile 0..31
        v8f acc = {};
        for (int ks = 0; ks < 32; ++ks) {
            // A operand: two 16-byte LDS chunks per lane
            union { uint4 q[2]; v16bf v; } au;
            const int kbase = ks << 5;
            au.q[0] = *(const uint4*)&smA[mrow * 1032 + kbase + 2 * koff];
            au.q[1] = *(const uint4*)&smA[mrow * 1032 + kbase + 2 * koff + 8];
            // B operand: one contiguous 32-byte load per lane
            v16bf bv = wp[(nt * 32 + ks) * 32 + lane];
            acc = __builtin_amdgcn_wmma_f32_16x16x32_bf16(
                false, au.v, false, bv, (short)0, acc, false, false);
        }
        // fused epilogue: h = tanh(acc + b1[col]); partial += h * W2[col]
        int col   = nt * 16 + (lane & 15);
        float b1v = b1[col];
        float w2v = W2[col];
#pragma unroll
        for (int r = 0; r < 8; ++r) {
            float h = tanhf(acc[r] + b1v);
            part[r] += h * w2v;
        }
    }

    // reduce partial logits across lanes/waves (LDS float atomics)
    const int mbase = (lane >> 4) << 3;
#pragma unroll
    for (int r = 0; r < 8; ++r)
        atomicAdd(&lpart[mbase + r], part[r]);
    __syncthreads();

    if (threadIdx.x < 16)
        logits[g0 + threadIdx.x] = lpart[threadIdx.x] + b2[0];
}

// ====================================================================
// Kernel 3: one block per (b,i) row.
//  - writes weights output row (sparse logits scatter)
//  - 5x gumbel-argmax -> OR into adj row, diagonal removed
// For rows i != num_nodes[b] the masked values are constant on valid
// columns, so argmax(masked+gumbel(u)) == argmax(u)  (monotone).
// ====================================================================
__global__ void __launch_bounds__(256)
rows_kernel(const float* __restrict__ logits,
            const int*   __restrict__ num_nodes,
            float* __restrict__ adj_out,
            float* __restrict__ w_out) {
    __shared__ float sv[256];
    __shared__ int   si[256];
    __shared__ int   winners[5];

    const int b  = blockIdx.x >> 9;
    const int i  = blockIdx.x & 511;
    const int nn = num_nodes[b];
    const int t  = threadIdx.x;
    const size_t rowoff = (size_t)blockIdx.x * NN;

    // ---- weights output row ----
#pragma unroll
    for (int p = 0; p < 2; ++p) {
        int j = t + p * 256;
        float wv = (i == nn && j < nn) ? logits[b * NN + j] : 0.0f;
        w_out[rowoff + j] = wv;
    }

    const bool rowvalid = (i <= nn);
    if (rowvalid) {
        const bool special = (i == nn);
        for (int e = 0; e < 5; ++e) {
            float bvv = -3.4e38f; int bj = 0;
#pragma unroll
            for (int p = 0; p < 2; ++p) {
                int j = t + p * 256;
                float z;
                if (j <= nn) {
                    float u = urand(e, b, i, j);
                    if (special) {
                        float mv = (j < nn) ? logits[b * NN + j] : 0.0f;
                        z = mv + (-__logf(-__logf(u)));   // masked + gumbel
                    } else {
                        z = u;                            // monotone surrogate
                    }
                } else {
                    z = -3.4e38f;
                }
                if (z > bvv) { bvv = z; bj = j; }
            }
            sv[t] = bvv; si[t] = bj;
            __syncthreads();
            for (int s = 128; s > 0; s >>= 1) {
                if (t < s && sv[t + s] > sv[t]) { sv[t] = sv[t + s]; si[t] = si[t + s]; }
                __syncthreads();
            }
            if (t == 0) winners[e] = si[0];
            __syncthreads();
        }
    }

    // ---- adj output row ----
#pragma unroll
    for (int p = 0; p < 2; ++p) {
        int j = t + p * 256;
        float a = 0.0f;
        if (rowvalid && j <= nn && j != i) {
#pragma unroll
            for (int e = 0; e < 5; ++e)
                if (winners[e] == j) a = 1.0f;
        }
        adj_out[rowoff + j] = a;
    }
}

// ====================================================================
extern "C" void kernel_launch(void* const* d_in, const int* in_sizes, int n_in,
                              void* d_out, int out_size, void* d_ws, size_t ws_size,
                              hipStream_t stream) {
    const float* nodes     = (const float*)d_in[0];
    // d_in[1] = adj (zeros), d_in[2] = weights (zeros) -- not needed
    const float* W1        = (const float*)d_in[3];
    const float* b1        = (const float*)d_in[4];
    const float* W2        = (const float*)d_in[5];
    const float* b2        = (const float*)d_in[6];
    const int*   num_nodes = (const int*)d_in[7];

    float* out     = (float*)d_out;
    float* adj_out = out;                                  // first tuple element
    float* w_out   = out + (size_t)BB * NN * NN;           // second tuple element

    unsigned short* W1p = (unsigned short*)d_ws;           // 1 MB packed bf16 W1
    float* logits = (float*)((char*)d_ws + (1u << 20));    // 128 KB logits

    w1pack_kernel<<<2048, 256, 0, stream>>>(W1, W1p);
    logits_kernel<<<(BB * NN) / 16, 256, 0, stream>>>(nodes, num_nodes, W1p,
                                                      b1, W2, b2, logits);
    rows_kernel<<<BB * NN, 256, 0, stream>>>(logits, num_nodes, adj_out, w_out);
}
